// SetAbstactionBlock_64458869178520
// MI455X (gfx1250) — matrix operand order repaired and checked by reading, not compiled, not run on hardware
//
#include <hip/hip_runtime.h>
#include <hip/hip_bf16.h>

// Problem constants (from reference)
#define BB 16
#define NN 4096
#define MM 1024
#define CC 64
#define KK 32
#define R2_ 0.04f
#define EPS_ 1e-5f

typedef __attribute__((ext_vector_type(16))) _Float16 v16h;
typedef __attribute__((ext_vector_type(8)))  _Float16 v8h;
typedef __attribute__((ext_vector_type(2)))  _Float16 v2h;
typedef __attribute__((ext_vector_type(8)))  float    v8f;

__device__ __forceinline__ float gelu_f(float x) {
    return 0.5f * x * (1.0f + erff(x * 0.70710678118654752440f));
}

// lane <-> lane^16 exchange via ds_swizzle SWAPX16 (xor=0x10, and=0x1f)
__device__ __forceinline__ float xor16(float v) {
    return __int_as_float(__builtin_amdgcn_ds_swizzle(__float_as_int(v), 0x401f));
}

// 16 consecutive halves -> one fragment register set (B fragment / generic)
__device__ __forceinline__ v16h ld16(const _Float16* p) {
    v8h lo = *(const v8h*)p;
    v8h hi = *(const v8h*)(p + 8);
    v16h r;
#pragma unroll
    for (int i = 0; i < 8; ++i) { r[i] = lo[i]; r[i + 8] = hi[i]; }
    return r;
}

// A fragment per ISA layout: runs [k0 .. k0+7] and [k0+16 .. k0+23] of this lane's row
__device__ __forceinline__ v16h ldA(const _Float16* row, int k0) {
    v8h lo = *(const v8h*)(row + k0);
    v8h hi = *(const v8h*)(row + k0 + 16);
    v16h r;
#pragma unroll
    for (int i = 0; i < 8; ++i) { r[i] = lo[i]; r[i + 8] = hi[i]; }
    return r;
}

__device__ __forceinline__ v8f wmma_f16(v16h a, v16h b, v8f c) {
    return __builtin_amdgcn_wmma_f32_16x16x32_f16(false, a, false, b, (short)0, c, false, false);
}

__device__ __forceinline__ unsigned ballot32(bool p) {
#if __has_builtin(__builtin_amdgcn_ballot_w32)
    return __builtin_amdgcn_ballot_w32(p);
#else
    return (unsigned)__ballot(p);
#endif
}

// ---------------------------------------------------------------------------
// prep: zero the stats block; convert weights to f16 (w1 column-permuted and
// zero-padded 67->96 to match feat layout [x(64) | cxyz(3) | 0...]).
// ---------------------------------------------------------------------------
__global__ void prep_kernel(const float* __restrict__ w1, const float* __restrict__ w2,
                            const float* __restrict__ w3, _Float16* __restrict__ w1h,
                            _Float16* __restrict__ w2h, _Float16* __restrict__ w3h,
                            float* __restrict__ stats) {
    int i = blockIdx.x * 256 + threadIdx.x;           // 32768 threads total
    if (i < 256 * 128) w3h[i] = (_Float16)w3[i];
    if (i < 128 * 64)  w2h[i] = (_Float16)w2[i];
    if (i < 64 * 96) {
        int o = i / 96, c = i % 96;
        float v = 0.0f;
        if (c < 64)      v = w1[o * 67 + 3 + c];      // feature cols
        else if (c < 67) v = w1[o * 67 + (c - 64)];   // xyz cols moved to the end
        w1h[i] = (_Float16)v;
    }
    if (i < 2048) stats[i] = 0.0f;
}

// ---------------------------------------------------------------------------
// Ball query: one wave per query; ordered first-K selection via ballot+prefix.
// ---------------------------------------------------------------------------
__global__ __launch_bounds__(256) void ballq_kernel(const float* __restrict__ src_xyz,
                                                    const float* __restrict__ xyz,
                                                    int* __restrict__ gidx) {
    const int wid = threadIdx.x >> 5, lane = threadIdx.x & 31;
    const int wq = blockIdx.x * 8 + wid;
    const int b = wq >> 10;                            // MM == 1024
    const float* q = xyz + (size_t)wq * 3;
    const float qx = q[0], qy = q[1], qz = q[2];
    const float* sp = src_xyz + (size_t)b * NN * 3;
    int* gout = gidx + (size_t)wq * KK;

    int cnt = 0, firstj = 0;
    bool havefirst = false;
    for (int base = 0; base < NN; base += 32) {
        const int j = base + lane;
        const float dx = sp[j * 3 + 0] - qx;
        const float dy = sp[j * 3 + 1] - qy;
        const float dz = sp[j * 3 + 2] - qz;
        const bool pred = (dx * dx + dy * dy + dz * dz) <= R2_;
        const unsigned mask = ballot32(pred);
        if (mask) {
            if (!havefirst) { firstj = base + __builtin_ctz(mask); havefirst = true; }
            const int prefix = __builtin_popcount(mask & ((1u << lane) - 1u));
            if (pred && (cnt + prefix) < KK) gout[cnt + prefix] = j;
            cnt += __builtin_popcount(mask);
            if (cnt >= KK) break;
        }
    }
    if (cnt < KK) {
        const int pad = havefirst ? firstj : 0;
        for (int i = cnt + lane; i < KK; i += 32) gout[i] = pad;
    }
}

// ---------------------------------------------------------------------------
// Fused per-query MLP pipeline. One wave per query; 4 waves / block.
// STAGE 1: gather + L1, emit L1 stats.
// STAGE 2: gather + L1(BN+GELU) + L2, emit L2 stats.
// STAGE 3: full pipeline + maxpool(K) -> h3max, emit final stats.
// LDS per wave: feat 32x104 (reused as h2 32x136 = 4352h), h1 32x72 (2304h).
// ---------------------------------------------------------------------------
template <int STAGE>
__global__ __launch_bounds__(128) void stage_kernel(
    const float* __restrict__ src_x, const float* __restrict__ src_xyz,
    const float* __restrict__ xyz, const int* __restrict__ gidx,
    const _Float16* __restrict__ w1h, const _Float16* __restrict__ w2h,
    const _Float16* __restrict__ w3h,
    const float* __restrict__ g1, const float* __restrict__ b1,
    const float* __restrict__ g2, const float* __restrict__ b2,
    const float* __restrict__ mean1, const float* __restrict__ istd1,
    const float* __restrict__ mean2, const float* __restrict__ istd2,
    float* __restrict__ gsum, float* __restrict__ gsq, float* __restrict__ h3max) {
    __shared__ __align__(16) _Float16 smem[4 * 6656];
    __shared__ float bsum[256], bsq[256];

    const int tid = threadIdx.x;
    const int wid = tid >> 5, lane = tid & 31;
    constexpr int NCH = (STAGE == 1) ? 64 : (STAGE == 2) ? 128 : 256;

    for (int i = tid; i < 256; i += 128) { bsum[i] = 0.0f; bsq[i] = 0.0f; }
    __syncthreads();

    const int wq = blockIdx.x * 4 + wid;
    const int b = wq >> 10;
    _Float16* feat = smem + wid * 6656;        // 32 rows x 104 (pad stride)
    _Float16* h1   = feat + 4352;              // 32 rows x 72
    _Float16* h2   = feat;                     // 32 rows x 136 (overlays feat)

    // ---- stage grouped features: row = neighbor (one per lane) ----
    {
        const int g = gidx[(size_t)wq * KK + lane];
        const float* sx = src_x + ((size_t)b * NN + g) * CC;
        _Float16* row = feat + lane * 104;
#pragma unroll
        for (int c = 0; c < CC; c += 4) {
            float4 v = *(const float4*)(sx + c);
            v2h h01 = {(_Float16)v.x, (_Float16)v.y};
            v2h h23 = {(_Float16)v.z, (_Float16)v.w};
            *(v2h*)(row + c) = h01;
            *(v2h*)(row + c + 2) = h23;
        }
        const float* p = src_xyz + ((size_t)b * NN + g) * 3;
        const float* q = xyz + (size_t)wq * 3;
        row[64] = (_Float16)(p[0] - q[0]);
        row[65] = (_Float16)(p[1] - q[1]);
        row[66] = (_Float16)(p[2] - q[2]);
#pragma unroll
        for (int c = 67; c < 96; ++c) row[c] = (_Float16)0.0f;
    }
    // same-wave LDS ops are in-order (DScnt): no barrier needed.

    const int hi8  = (lane & 16) >> 1;   // A-fragment K sub-offset per lane half
    const int hi16 = (lane & 16);        // unused: B fragments use hi8 as their K sub-offset
    const int col  = lane & 15;

    // ---- layer 1: (32x96) @ (96x64) ----
#pragma unroll
    for (int ct = 0; ct < 4; ++ct) {
        v8f acc0 = {0, 0, 0, 0, 0, 0, 0, 0}, acc1 = {0, 0, 0, 0, 0, 0, 0, 0};
        const _Float16* wrow = w1h + (ct * 16 + col) * 96 + hi8;
#pragma unroll
        for (int ks = 0; ks < 3; ++ks) {
            v16h a0 = ldA(feat + col * 104, ks * 32 + hi8);
            v16h a1 = ldA(feat + (col + 16) * 104, ks * 32 + hi8);
            v16h bf = ldA(wrow, ks * 32);
            acc0 = wmma_f16(a0, bf, acc0);
            acc1 = wmma_f16(a1, bf, acc1);
        }
        if constexpr (STAGE == 1) {
            float s = 0.0f, s2 = 0.0f;
#pragma unroll
            for (int r = 0; r < 8; ++r) {
                s  += acc0[r] + acc1[r];
                s2 += acc0[r] * acc0[r] + acc1[r] * acc1[r];
            }
            s += xor16(s); s2 += xor16(s2);
            if (lane < 16) { atomicAdd(&bsum[ct * 16 + lane], s); atomicAdd(&bsq[ct * 16 + lane], s2); }
        } else {
            const int ch = ct * 16 + col;
            const float sc = istd1[ch] * g1[ch];
            const float sh = b1[ch] - mean1[ch] * sc;
#pragma unroll
            for (int r = 0; r < 8; ++r) {
                h1[(r + hi8) * 72 + ch]      = (_Float16)gelu_f(acc0[r] * sc + sh);
                h1[(16 + r + hi8) * 72 + ch] = (_Float16)gelu_f(acc1[r] * sc + sh);
            }
        }
    }

    // ---- layer 2: (32x64) @ (64x128) ----
    if constexpr (STAGE >= 2) {
#pragma unroll
        for (int ct = 0; ct < 8; ++ct) {
            v8f acc0 = {0, 0, 0, 0, 0, 0, 0, 0}, acc1 = {0, 0, 0, 0, 0, 0, 0, 0};
            const _Float16* wrow = w2h + (ct * 16 + col) * 64 + hi8;
#pragma unroll
            for (int ks = 0; ks < 2; ++ks) {
                v16h a0 = ldA(h1 + col * 72, ks * 32 + hi8);
                v16h a1 = ldA(h1 + (col + 16) * 72, ks * 32 + hi8);
                v16h bf = ldA(wrow, ks * 32);
                acc0 = wmma_f16(a0, bf, acc0);
                acc1 = wmma_f16(a1, bf, acc1);
            }
            if constexpr (STAGE == 2) {
                float s = 0.0f, s2 = 0.0f;
#pragma unroll
                for (int r = 0; r < 8; ++r) {
                    s  += acc0[r] + acc1[r];
                    s2 += acc0[r] * acc0[r] + acc1[r] * acc1[r];
                }
                s += xor16(s); s2 += xor16(s2);
                if (lane < 16) { atomicAdd(&bsum[ct * 16 + lane], s); atomicAdd(&bsq[ct * 16 + lane], s2); }
            } else {
                const int ch = ct * 16 + col;
                const float sc = istd2[ch] * g2[ch];
                const float sh = b2[ch] - mean2[ch] * sc;
#pragma unroll
                for (int r = 0; r < 8; ++r) {
                    h2[(r + hi8) * 136 + ch]      = (_Float16)gelu_f(acc0[r] * sc + sh);
                    h2[(16 + r + hi8) * 136 + ch] = (_Float16)gelu_f(acc1[r] * sc + sh);
                }
            }
        }
    }

    // ---- layer 3: (32x128) @ (128x256) + maxpool over K ----
    if constexpr (STAGE == 3) {
        for (int ct = 0; ct < 16; ++ct) {
            v8f acc0 = {0, 0, 0, 0, 0, 0, 0, 0}, acc1 = {0, 0, 0, 0, 0, 0, 0, 0};
            const _Float16* wrow = w3h + (ct * 16 + col) * 128 + hi8;
#pragma unroll
            for (int ks = 0; ks < 4; ++ks) {
                v16h a0 = ldA(h2 + col * 136, ks * 32 + hi8);
                v16h a1 = ldA(h2 + (col + 16) * 136, ks * 32 + hi8);
                v16h bf = ldA(wrow, ks * 32);
                acc0 = wmma_f16(a0, bf, acc0);
                acc1 = wmma_f16(a1, bf, acc1);
            }
            float mx = acc0[0];
#pragma unroll
            for (int r = 0; r < 8; ++r) { mx = fmaxf(mx, acc0[r]); mx = fmaxf(mx, acc1[r]); }
            mx = fmaxf(mx, xor16(mx));
            if (lane < 16) {
                const int ch = ct * 16 + lane;
                h3max[(size_t)wq * 256 + ch] = mx;
                atomicAdd(&bsum[ch], mx);
                atomicAdd(&bsq[ch], mx * mx);
            }
        }
    }

    // ---- flush block-local stats ----
    __syncthreads();
    for (int i = tid; i < NCH; i += 128) {
        atomicAdd(&gsum[i], bsum[i]);
        atomicAdd(&gsq[i], bsq[i]);
    }
}

__global__ void finalize_kernel(const float* __restrict__ sum, const float* __restrict__ sq,
                                float* __restrict__ mean, float* __restrict__ istd,
                                int n, float inv_cnt) {
    int i = blockIdx.x * blockDim.x + threadIdx.x;
    if (i < n) {
        const float m = sum[i] * inv_cnt;
        const float v = sq[i] * inv_cnt - m * m;
        mean[i] = m;
        istd[i] = rsqrtf(v + EPS_);
    }
}

__global__ void final_out_kernel(const float* __restrict__ h3max, const float* __restrict__ meanL,
                                 const float* __restrict__ istdL, const float* __restrict__ gl,
                                 const float* __restrict__ bl, float* __restrict__ out) {
    const size_t i = (size_t)blockIdx.x * 256 + threadIdx.x;
    const int ch = (int)(i & 255);
    const float sc = istdL[ch] * gl[ch];
    const float sh = bl[ch] - meanL[ch] * sc;
    out[i] = gelu_f(h3max[i] * sc + sh);
}

extern "C" void kernel_launch(void* const* d_in, const int* in_sizes, int n_in,
                              void* d_out, int out_size, void* d_ws, size_t ws_size,
                              hipStream_t stream) {
    (void)in_sizes; (void)n_in; (void)out_size; (void)ws_size;
    const float* src_x   = (const float*)d_in[0];
    const float* src_xyz = (const float*)d_in[1];
    const float* xyz     = (const float*)d_in[2];
    const float* w1 = (const float*)d_in[3];
    const float* g1 = (const float*)d_in[4];
    const float* b1 = (const float*)d_in[5];
    const float* w2 = (const float*)d_in[6];
    const float* g2 = (const float*)d_in[7];
    const float* b2 = (const float*)d_in[8];
    const float* w3 = (const float*)d_in[9];
    const float* gl = (const float*)d_in[10];
    const float* bl = (const float*)d_in[11];
    float* out = (float*)d_out;

    // Workspace layout (~18.1 MB total)
    char* ws = (char*)d_ws;
    int*      gidx  = (int*)ws;                                      // 2 MB
    _Float16* w1h   = (_Float16*)(ws + 2097152);                     // 12 KB
    _Float16* w2h   = (_Float16*)(ws + 2097152 + 12288);             // 16 KB
    _Float16* w3h   = (_Float16*)(ws + 2097152 + 12288 + 16384);     // 64 KB
    float*    stats = (float*)(ws + 2097152 + 12288 + 16384 + 65536);// 8 KB
    float*    h3max = (float*)(ws + 2097152 + 12288 + 16384 + 65536 + 8192); // 16 MB

    float *sum1 = stats +    0, *sq1 = stats +   64, *mean1 = stats +  128, *istd1 = stats +  192;
    float *sum2 = stats +  256, *sq2 = stats +  384, *mean2 = stats +  512, *istd2 = stats +  640;
    float *sumL = stats +  768, *sqL = stats + 1024, *meanL = stats + 1280, *istdL = stats + 1536;

    prep_kernel<<<128, 256, 0, stream>>>(w1, w2, w3, w1h, w2h, w3h, stats);
    ballq_kernel<<<BB * MM / 8, 256, 0, stream>>>(src_xyz, xyz, gidx);

    stage_kernel<1><<<BB * MM / 4, 128, 0, stream>>>(src_x, src_xyz, xyz, gidx, w1h, w2h, w3h,
                                                     g1, b1, g2, b2, mean1, istd1, mean2, istd2,
                                                     sum1, sq1, h3max);
    finalize_kernel<<<1, 256, 0, stream>>>(sum1, sq1, mean1, istd1, 64, 1.0f / (BB * MM * KK));

    stage_kernel<2><<<BB * MM / 4, 128, 0, stream>>>(src_x, src_xyz, xyz, gidx, w1h, w2h, w3h,
                                                     g1, b1, g2, b2, mean1, istd1, mean2, istd2,
                                                     sum2, sq2, h3max);
    finalize_kernel<<<1, 256, 0, stream>>>(sum2, sq2, mean2, istd2, 128, 1.0f / (BB * MM * KK));

    stage_kernel<3><<<BB * MM / 4, 128, 0, stream>>>(src_x, src_xyz, xyz, gidx, w1h, w2h, w3h,
                                                     g1, b1, g2, b2, mean1, istd1, mean2, istd2,
                                                     sumL, sqL, h3max);
    finalize_kernel<<<1, 256, 0, stream>>>(sumL, sqL, meanL, istdL, 256, 1.0f / (BB * MM));

    final_out_kernel<<<BB * MM, 256, 0, stream>>>(h3max, meanL, istdL, gl, bl, out);
}
